// GenerativeGNN_7937099563262
// MI455X (gfx1250) — matrix operand order repaired and checked by reading, not compiled, not run on hardware
//
#include <hip/hip_runtime.h>
#include <hip/hip_bf16.h>
#include <stdint.h>

#define NN 50000
#define NE 800000

typedef _Float16 half4 __attribute__((ext_vector_type(4)));
typedef _Float16 half8 __attribute__((ext_vector_type(8)));
typedef _Float16 v16h  __attribute__((ext_vector_type(16)));
typedef float    v8f   __attribute__((ext_vector_type(8)));

// ---------------- fragment loaders (per ISA 7.12.2 layouts, wave32) ----------
// A 16x32 f16: lane<16 -> row=lane, halves[0:8]=K+0..7, halves[8:16]=K+16..23
//              lane>=16 -> row=lane-16, halves = K+8..15 / K+24..31
__device__ __forceinline__ v16h frag_a(const _Float16* p) {
  half8 lo = *(const half8*)(p);
  half8 hi = *(const half8*)(p + 16);
  return __builtin_shufflevector(lo, hi, 0,1,2,3,4,5,6,7,8,9,10,11,12,13,14,15);
}
// B 32x16 f16 (from col-major W^T, row stride K): halves K+0..7 / K+16..23 (+8 hi lanes)
__device__ __forceinline__ v16h frag_b(const _Float16* p) {
  half8 lo = *(const half8*)(p);
  half8 hi = *(const half8*)(p + 16);
  return __builtin_shufflevector(lo, hi, 0,1,2,3,4,5,6,7,8,9,10,11,12,13,14,15);
}

// ---------------- elementwise / sparse kernels ------------------------------

// x / 1000 - 0.5 -> f16 (vec4)
__global__ void k_prep(const float4* __restrict__ x, half4* __restrict__ out, int total4) {
  int t = blockIdx.x * blockDim.x + threadIdx.x;
  if (t >= total4) return;
  float4 v = x[t];
  half4 h;
  h[0] = (_Float16)(v.x * 0.001f - 0.5f);
  h[1] = (_Float16)(v.y * 0.001f - 0.5f);
  h[2] = (_Float16)(v.z * 0.001f - 0.5f);
  h[3] = (_Float16)(v.w * 0.001f - 0.5f);
  out[t] = h;
}

// W (K x Ho, f32 row-major) -> Wt (Ho x K, f16 row-major)  [transpose+convert]
__global__ void k_wconv(const float* __restrict__ W, _Float16* __restrict__ Wt, int K, int Ho) {
  int t = blockIdx.x * blockDim.x + threadIdx.x;
  if (t >= K * Ho) return;
  int k = t / Ho, n = t % Ho;
  Wt[(size_t)n * K + k] = (_Float16)W[t];
}

__global__ void k_count(const int* __restrict__ dst, float* __restrict__ cnt, int e) {
  int t = blockIdx.x * blockDim.x + threadIdx.x;
  if (t >= e) return;
  atomicAdd(&cnt[dst[t]], 1.0f);
}

__global__ void k_inv(const float* __restrict__ cnt, float* __restrict__ inv, int n) {
  int t = blockIdx.x * blockDim.x + threadIdx.x;
  if (t >= n) return;
  inv[t] = 1.0f / fmaxf(cnt[t], 1.0f);
}

// scatter-add messages: agg[dst] += X[src]; 8 features per thread
__global__ void k_scatter(const _Float16* __restrict__ X, const int* __restrict__ src,
                          const int* __restrict__ dst, float* __restrict__ agg,
                          int K, int chunkShift, long long total) {
  long long t = (long long)blockIdx.x * blockDim.x + threadIdx.x;
  if (t >= total) return;
  int e = (int)(t >> chunkShift);
  int f = ((int)t & ((1 << chunkShift) - 1)) << 3;
  int s = src[e], d = dst[e];
  half8 v = *(const half8*)(X + (size_t)s * K + f);
  float* base = agg + (size_t)d * K + f;
#pragma unroll
  for (int i = 0; i < 8; i++) atomicAdd(base + i, (float)v[i]);
}

// mean = agg * inv_deg -> f16 (vec4)
__global__ void k_mean(const float4* __restrict__ agg, const float* __restrict__ inv,
                       half4* __restrict__ mean, int shiftK, int total4) {
  int t = blockIdx.x * blockDim.x + threadIdx.x;
  if (t >= total4) return;
  int row = (int)(((long long)t * 4) >> shiftK);
  float4 a = agg[t];
  float s = inv[row];
  half4 h;
  h[0] = (_Float16)(a.x * s); h[1] = (_Float16)(a.y * s);
  h[2] = (_Float16)(a.z * s); h[3] = (_Float16)(a.w * s);
  mean[t] = h;
}

// z = mean + exp(log_var) * eps  -> f16 (vec4)
__global__ void k_reparam(const float4* __restrict__ mean, const float4* __restrict__ lv,
                          const float4* __restrict__ eps, half4* __restrict__ z, int total4) {
  int t = blockIdx.x * blockDim.x + threadIdx.x;
  if (t >= total4) return;
  float4 m = mean[t], l = lv[t], e = eps[t];
  half4 h;
  h[0] = (_Float16)(m.x + __expf(l.x) * e.x);
  h[1] = (_Float16)(m.y + __expf(l.y) * e.y);
  h[2] = (_Float16)(m.z + __expf(l.z) * e.z);
  h[3] = (_Float16)(m.w + __expf(l.w) * e.w);
  z[t] = h;
}

// ---------------- WMMA GEMM: D = act(A1@W1t + [A2@W2t] + bias) --------------
// one wave computes a 16x64 output tile (4 accumulators, A-fragment reused 4x)
// 8 waves per block; wave32; all branches wave-uniform so EXEC stays all-ones
__global__ void k_gemm(const _Float16* __restrict__ A1, const _Float16* __restrict__ W1,
                       const _Float16* __restrict__ A2, const _Float16* __restrict__ W2,
                       const float* __restrict__ bias,
                       int K, int Ho, int numGroupsN, int totalTiles, int act,
                       _Float16* __restrict__ outH,
                       float* __restrict__ outF, float* __restrict__ outF2, int splitCol) {
  int wave = threadIdx.x >> 5;
  int lane = threadIdx.x & 31;
  int tile = blockIdx.x * (blockDim.x >> 5) + wave;
  if (tile >= totalTiles) return;                 // wave-uniform
  int mTile  = tile / numGroupsN;
  int nGroup = tile - mTile * numGroupsN;
  int colB = nGroup * 64 + (lane & 15);           // column for B frag / bias / store
  int rowA = mTile * 16 + (lane & 15);
  int aOff = (lane >> 4) * 8;
  int bOff = (lane >> 4) * 8;

  v8f acc[4];
#pragma unroll
  for (int c = 0; c < 4; c++) {
    float bv = bias[colB + c * 16];
#pragma unroll
    for (int i = 0; i < 8; i++) acc[c][i] = bv;
  }

  const _Float16* a1p = A1 + (size_t)rowA * K + aOff;
  const _Float16* b1p = W1 + (size_t)colB * K + bOff;   // +c*16*K per column tile
  for (int k0 = 0; k0 < K; k0 += 32) {
    v16h a = frag_a(a1p + k0);
#pragma unroll
    for (int c = 0; c < 4; c++) {
      v16h b = frag_b(b1p + (size_t)c * 16 * K + k0);
      acc[c] = __builtin_amdgcn_wmma_f32_16x16x32_f16(false, a, false, b, (short)0, acc[c],
                                                      false, false);
    }
  }
  if (A2) {
    const _Float16* a2p = A2 + (size_t)rowA * K + aOff;
    const _Float16* b2p = W2 + (size_t)colB * K + bOff;
    for (int k0 = 0; k0 < K; k0 += 32) {
      v16h a = frag_a(a2p + k0);
#pragma unroll
      for (int c = 0; c < 4; c++) {
        v16h b = frag_b(b2p + (size_t)c * 16 * K + k0);
        acc[c] = __builtin_amdgcn_wmma_f32_16x16x32_f16(false, a, false, b, (short)0, acc[c],
                                                        false, false);
      }
    }
  }

  int rBase = mTile * 16 + ((lane >> 4) * 8);     // C/D layout: VGPR r -> row r (+8 hi lanes)
#pragma unroll
  for (int c = 0; c < 4; c++) {
    int col = colB + c * 16;
#pragma unroll
    for (int r = 0; r < 8; r++) {
      int row = rBase + r;
      float v = acc[c][r];
      if (act == 1) v = __sinf(v);
      else if (act == 2) v = fmaxf(v, 0.0f);
      else if (act == 3) v = 1000.0f / (1.0f + __expf(-v));
      if (outH) outH[(size_t)row * Ho + col] = (_Float16)v;
      if (outF) {
        if (outF2 && col >= splitCol) outF2[(size_t)row * splitCol + (col - splitCol)] = v;
        else                          outF [(size_t)row * splitCol + col] = v;
      }
    }
  }
}

// ---------------- host-side orchestration -----------------------------------

static inline size_t alignup(size_t x) { return (x + 255) & ~(size_t)255; }

extern "C" void kernel_launch(void* const* d_in, const int* in_sizes, int n_in,
                              void* d_out, int out_size, void* d_ws, size_t ws_size,
                              hipStream_t stream) {
  const float* x    = (const float*)d_in[0];
  const int*   src  = (const int*)d_in[1];
  const int*   dst  = src + NE;
  const float* eps  = (const float*)d_in[2];
  const float* Wl1  = (const float*)d_in[3];  const float* bl1 = (const float*)d_in[4];
  const float* Wr1  = (const float*)d_in[5];
  const float* Wl2  = (const float*)d_in[6];  const float* bl2 = (const float*)d_in[7];
  const float* Wr2  = (const float*)d_in[8];
  const float* Wl3  = (const float*)d_in[9];  const float* bl3 = (const float*)d_in[10];
  const float* Wr3  = (const float*)d_in[11];
  const float* Wl4  = (const float*)d_in[12]; const float* bl4 = (const float*)d_in[13];
  const float* Wr4  = (const float*)d_in[14];
  const float* Wlin = (const float*)d_in[15]; const float* blin = (const float*)d_in[16];

  float* outF    = (float*)d_out;              // N x 64
  float* meanOut = outF + (size_t)NN * 64;     // N x 128
  float* lvOut   = meanOut + (size_t)NN * 128; // N x 128

  char* ws = (char*)d_ws; size_t off = 0;
  float*    cnt   = (float*)(ws + off);    off += alignup(NN * 4);
  float*    inv   = (float*)(ws + off);    off += alignup(NN * 4);
  float*    agg   = (float*)(ws + off);    off += alignup((size_t)NN * 256 * 4);
  _Float16* bufA  = (_Float16*)(ws + off); off += alignup((size_t)NN * 256 * 2);
  _Float16* bufB  = (_Float16*)(ws + off); off += alignup((size_t)NN * 256 * 2);
  _Float16* meanF = (_Float16*)(ws + off); off += alignup((size_t)NN * 256 * 2);
  _Float16* Wl1t  = (_Float16*)(ws + off); off += alignup(128 * 256 * 2);
  _Float16* Wr1t  = (_Float16*)(ws + off); off += alignup(128 * 256 * 2);
  _Float16* Wl2t  = (_Float16*)(ws + off); off += alignup(256 * 256 * 2);
  _Float16* Wr2t  = (_Float16*)(ws + off); off += alignup(256 * 256 * 2);
  _Float16* Wl3t  = (_Float16*)(ws + off); off += alignup(128 * 128 * 2);
  _Float16* Wr3t  = (_Float16*)(ws + off); off += alignup(128 * 128 * 2);
  _Float16* Wl4t  = (_Float16*)(ws + off); off += alignup(128 * 128 * 2);
  _Float16* Wr4t  = (_Float16*)(ws + off); off += alignup(128 * 128 * 2);
  _Float16* Wlint = (_Float16*)(ws + off); off += alignup(128 * 64 * 2);

  const int B = 256;
#define GRID(n) ((int)(((n) + B - 1) / B))

  // weight convert+transpose (f32 -> f16, col-major)
  k_wconv<<<GRID(128 * 256), B, 0, stream>>>(Wl1, Wl1t, 128, 256);
  k_wconv<<<GRID(128 * 256), B, 0, stream>>>(Wr1, Wr1t, 128, 256);
  k_wconv<<<GRID(256 * 256), B, 0, stream>>>(Wl2, Wl2t, 256, 256);
  k_wconv<<<GRID(256 * 256), B, 0, stream>>>(Wr2, Wr2t, 256, 256);
  k_wconv<<<GRID(128 * 128), B, 0, stream>>>(Wl3, Wl3t, 128, 128);
  k_wconv<<<GRID(128 * 128), B, 0, stream>>>(Wr3, Wr3t, 128, 128);
  k_wconv<<<GRID(128 * 128), B, 0, stream>>>(Wl4, Wl4t, 128, 128);
  k_wconv<<<GRID(128 * 128), B, 0, stream>>>(Wr4, Wr4t, 128, 128);
  k_wconv<<<GRID(128 * 64),  B, 0, stream>>>(Wlin, Wlint, 128, 64);

  // h0 = x/1000 - 0.5 (f16)
  k_prep<<<GRID(NN * 128 / 4), B, 0, stream>>>((const float4*)x, (half4*)bufA, NN * 128 / 4);

  // degrees (shared by all layers)
  hipMemsetAsync(cnt, 0, NN * 4, stream);
  k_count<<<GRID(NE), B, 0, stream>>>(dst, cnt, NE);
  k_inv<<<GRID(NN), B, 0, stream>>>(cnt, inv, NN);

  const int wavesPerBlock = 8;
  long long tot;

  // ---- layer 1: K=128 -> H=256, sin ----
  hipMemsetAsync(agg, 0, (size_t)NN * 128 * 4, stream);
  tot = (long long)NE * (128 / 8);
  k_scatter<<<GRID(tot), B, 0, stream>>>(bufA, src, dst, agg, 128, 4, tot);
  k_mean<<<GRID(NN * 128 / 4), B, 0, stream>>>((const float4*)agg, inv, (half4*)meanF, 7, NN * 128 / 4);
  {
    int groups = 256 / 64, tiles = (NN / 16) * groups;
    k_gemm<<<(tiles + wavesPerBlock - 1) / wavesPerBlock, 32 * wavesPerBlock, 0, stream>>>(
        meanF, Wl1t, bufA, Wr1t, bl1, 128, 256, groups, tiles, /*sin*/1,
        bufB, nullptr, nullptr, 256);
  }

  // ---- layer 2: K=256 -> H=256, split mean/logvar straight to d_out ----
  hipMemsetAsync(agg, 0, (size_t)NN * 256 * 4, stream);
  tot = (long long)NE * (256 / 8);
  k_scatter<<<GRID(tot), B, 0, stream>>>(bufB, src, dst, agg, 256, 5, tot);
  k_mean<<<GRID(NN * 256 / 4), B, 0, stream>>>((const float4*)agg, inv, (half4*)meanF, 8, NN * 256 / 4);
  {
    int groups = 256 / 64, tiles = (NN / 16) * groups;
    k_gemm<<<(tiles + wavesPerBlock - 1) / wavesPerBlock, 32 * wavesPerBlock, 0, stream>>>(
        meanF, Wl2t, bufB, Wr2t, bl2, 256, 256, groups, tiles, /*none*/0,
        nullptr, meanOut, lvOut, 128);
  }

  // z = mean + exp(log_var)*eps (f16)
  k_reparam<<<GRID(NN * 128 / 4), B, 0, stream>>>((const float4*)meanOut, (const float4*)lvOut,
                                                  (const float4*)eps, (half4*)bufA, NN * 128 / 4);

  // ---- layer 3: K=128 -> 128, relu ----
  hipMemsetAsync(agg, 0, (size_t)NN * 128 * 4, stream);
  tot = (long long)NE * (128 / 8);
  k_scatter<<<GRID(tot), B, 0, stream>>>(bufA, src, dst, agg, 128, 4, tot);
  k_mean<<<GRID(NN * 128 / 4), B, 0, stream>>>((const float4*)agg, inv, (half4*)meanF, 7, NN * 128 / 4);
  {
    int groups = 128 / 64, tiles = (NN / 16) * groups;
    k_gemm<<<(tiles + wavesPerBlock - 1) / wavesPerBlock, 32 * wavesPerBlock, 0, stream>>>(
        meanF, Wl3t, bufA, Wr3t, bl3, 128, 128, groups, tiles, /*relu*/2,
        bufB, nullptr, nullptr, 128);
  }

  // ---- layer 4: K=128 -> 128, relu ----
  hipMemsetAsync(agg, 0, (size_t)NN * 128 * 4, stream);
  k_scatter<<<GRID(tot), B, 0, stream>>>(bufB, src, dst, agg, 128, 4, tot);
  k_mean<<<GRID(NN * 128 / 4), B, 0, stream>>>((const float4*)agg, inv, (half4*)meanF, 7, NN * 128 / 4);
  {
    int groups = 128 / 64, tiles = (NN / 16) * groups;
    k_gemm<<<(tiles + wavesPerBlock - 1) / wavesPerBlock, 32 * wavesPerBlock, 0, stream>>>(
        meanF, Wl4t, bufB, Wr4t, bl4, 128, 128, groups, tiles, /*relu*/2,
        bufA, nullptr, nullptr, 128);
  }

  // ---- final: K=128 -> 64, sigmoid*1000, f32 out ----
  {
    int groups = 64 / 64, tiles = (NN / 16) * groups;
    k_gemm<<<(tiles + wavesPerBlock - 1) / wavesPerBlock, 32 * wavesPerBlock, 0, stream>>>(
        bufA, Wlint, nullptr, nullptr, blin, 128, 64, groups, tiles, /*sigmoid*1000*/3,
        nullptr, outF, nullptr, 64);
  }
#undef GRID
}
